// QKVAttention_2405181686216
// MI455X (gfx1250) — matrix operand order repaired and checked by reading, not compiled, not run on hardware
//
#include <hip/hip_runtime.h>
#include <hip/hip_bf16.h>

typedef __attribute__((ext_vector_type(16))) __bf16 v16bf;
typedef __attribute__((ext_vector_type(8)))  __bf16 v8bf;
typedef __attribute__((ext_vector_type(4)))  __bf16 v4bf;
typedef __attribute__((ext_vector_type(8)))  float  v8f;

#define SEQ_T  2048
#define DH     64
#define NBH    32   // bs*heads

__device__ __forceinline__ unsigned pack2(float lo, float hi) {
    __bf16 a = (__bf16)lo, b = (__bf16)hi;
    unsigned short ua = __builtin_bit_cast(unsigned short, a);
    unsigned short ub = __builtin_bit_cast(unsigned short, b);
    return (unsigned)ua | ((unsigned)ub << 16);
}

// CDNA5 async copy: 16 bytes global -> LDS, tracked by ASYNCcnt (GVS mode).
__device__ __forceinline__ void async_b128(unsigned ldsOff, unsigned gOff, const void* base) {
    asm volatile("global_load_async_to_lds_b128 %0, %1, %2"
                 :: "v"(ldsOff), "v"(gOff), "s"(base) : "memory");
}
__device__ __forceinline__ void wait_async0() {
    asm volatile("s_wait_asynccnt 0" ::: "memory");
}

// ---------------------------------------------------------------------------
// Pre-pass 1: Q/K  f32 (bh, c, t) -> bf16 transposed (bh, t, c), with scale.
// ---------------------------------------------------------------------------
__global__ __launch_bounds__(256) void qk_transpose(const float* __restrict__ qkv,
                                                    __bf16* __restrict__ dstQ,
                                                    __bf16* __restrict__ dstK) {
    __shared__ __bf16 tile[64][66];
    const int sect = blockIdx.y;              // 0 = Q, 1 = K
    const int bh   = blockIdx.x >> 5;         // 0..31
    const int t0   = (blockIdx.x & 31) << 6;  // 64-wide t slice
    const int b = bh >> 3, h = bh & 7;
    const float scale = 0.35355339059327373f; // 64^-0.25
    const float* src = qkv + ((size_t)b * 1536 + (size_t)sect * 512 + (size_t)h * 64) * SEQ_T + t0;
    const int tid = threadIdx.x;
#pragma unroll
    for (int i = 0; i < 16; ++i) {
        int idx = tid + 256 * i;
        int c = idx >> 6, tt = idx & 63;
        tile[c][tt] = (__bf16)(src[(size_t)c * SEQ_T + tt] * scale);
    }
    __syncthreads();
    __bf16* dst = (sect ? dstK : dstQ) + ((size_t)bh * SEQ_T + t0) * DH;
#pragma unroll
    for (int i = 0; i < 16; ++i) {
        int idx = tid + 256 * i;
        int tt = idx >> 6, c = idx & 63;
        dst[(size_t)tt * DH + c] = tile[c][tt];
    }
}

// ---------------------------------------------------------------------------
// Pre-pass 2: V f32 -> bf16, same (bh, c, s) layout.
// ---------------------------------------------------------------------------
__global__ __launch_bounds__(256) void v_convert(const float* __restrict__ qkv,
                                                 __bf16* __restrict__ dstV) {
    size_t i = (size_t)blockIdx.x * blockDim.x + threadIdx.x; // float4 index
    size_t e = i * 4;
    size_t row = e >> 11;          // bh*64 + c
    size_t s   = e & 2047;
    size_t bh = row >> 6, c = row & 63;
    size_t b = bh >> 3, h = bh & 7;
    const float4 f = *(const float4*)(qkv + ((b * 1536 + 1024 + h * 64 + c) * (size_t)SEQ_T + s));
    v4bf o; o.x = (__bf16)f.x; o.y = (__bf16)f.y; o.z = (__bf16)f.z; o.w = (__bf16)f.w;
    *(v4bf*)(dstV + e) = o;
}

// ---------------------------------------------------------------------------
// Flash attention. Block = 8 wave32, each wave owns 16 query rows.
// K/V tiles double-buffered in LDS via async copies (ASYNCcnt).
// ---------------------------------------------------------------------------
__global__ __launch_bounds__(256) void flash_attn(const __bf16* __restrict__ Qb,
                                                  const __bf16* __restrict__ Kb,
                                                  const __bf16* __restrict__ Vb,
                                                  float* __restrict__ out) {
    // Padded LDS tiles: K 32x64 (rows padded to 72 elems = 144B, 16B-aligned),
    // V 64x32 (rows padded to 40 elems = 80B, 16B-aligned).
    __shared__ __align__(16) __bf16 kT[2][32][72];
    __shared__ __align__(16) __bf16 vT[2][64][40];

    const int tid  = threadIdx.x;
    const int lane = tid & 31;
    const int ln   = lane & 15;
    const int hi   = lane >> 4;
    const int w    = tid >> 5;
    const int bh   = blockIdx.y;
    const int tq   = blockIdx.x * 128 + w * 16 + ln;  // this lane's query column

    const __bf16* Qp    = Qb + ((size_t)bh * SEQ_T + tq) * DH;
    const __bf16* Kbase = Kb + (size_t)bh * SEQ_T * DH;
    const __bf16* Vbase = Vb + (size_t)bh * DH * SEQ_T;

    // Per-thread async-copy geometry (byte offsets).
    const unsigned ldsK0 = (unsigned)(uintptr_t)&kT[0][0][0];
    const unsigned ldsV0 = (unsigned)(uintptr_t)&vT[0][0][0];
    const int krow = tid >> 3, kcolB = (tid & 7) * 16;  // 32 rows x 128B
    const int vrow = tid >> 2, vcolB = (tid & 3) * 16;  // 64 rows x 64B
    const unsigned kLdsOff = ldsK0 + (unsigned)(krow * 144 + kcolB);
    const unsigned vLdsOff = ldsV0 + (unsigned)(vrow * 80 + vcolB);
    const unsigned kGOff   = (unsigned)(krow * (DH * 2) + kcolB);
    const unsigned vGOff   = (unsigned)(vrow * (SEQ_T * 2) + vcolB);

    // Q as WMMA B-operand (K-dim = c). Lane holds column t, rows c = 32f+16*hi+0..15.
    union U16 { v16bf v; v8bf h[2]; };
    v16bf qf0 = *(const v16bf*)(Qp + 16 * hi);
    v16bf qf1 = *(const v16bf*)(Qp + 32 + 16 * hi);

    v8f acc[4] = {v8f{}, v8f{}, v8f{}, v8f{}};
    float m = -3.0e38f, l = 0.f;
    const v8f zero{};

    // Prologue: stage tile 0 into buffer 0.
    async_b128(kLdsOff, kGOff, Kbase);
    async_b128(vLdsOff, vGOff + 0, Vbase);
    wait_async0();
    __syncthreads();

    for (int step = 0; step < SEQ_T / 32; ++step) {
        const int cur = step & 1;
        const int s0  = step * 32;

        // Overlap: stage next tile into the other buffer while computing.
        if (step + 1 < SEQ_T / 32) {
            const unsigned buf = (unsigned)(1 - cur);
            async_b128(kLdsOff + buf * (32u * 144u), kGOff + (unsigned)(s0 + 32) * (DH * 2), Kbase);
            async_b128(vLdsOff + buf * (64u * 80u),  vGOff + (unsigned)(s0 + 32) * 2,        Vbase);
        }

        // ---- Sᵀ = Kᵀ(16s x 32c) x Q(32c x 16t), two s-row tiles -------------
        U16 ka[2][2];
#pragma unroll
        for (int g = 0; g < 2; ++g) {
#pragma unroll
            for (int f = 0; f < 2; ++f) {
                ka[g][f].h[0] = *(const v8bf*)&kT[cur][16 * g + ln][32 * f + 16 * hi];
                ka[g][f].h[1] = *(const v8bf*)&kT[cur][16 * g + ln][32 * f + 16 * hi + 8];
            }
        }
        v8f d0 = __builtin_amdgcn_wmma_f32_16x16x32_bf16(false, ka[0][0].v, false, qf0, (short)0, zero, false, false);
        d0     = __builtin_amdgcn_wmma_f32_16x16x32_bf16(false, ka[0][1].v, false, qf1, (short)0, d0,   false, false);
        v8f d1 = __builtin_amdgcn_wmma_f32_16x16x32_bf16(false, ka[1][0].v, false, qf0, (short)0, zero, false, false);
        d1     = __builtin_amdgcn_wmma_f32_16x16x32_bf16(false, ka[1][1].v, false, qf1, (short)0, d1,   false, false);

        // ---- online softmax: lane's column t is fixed -> scalar stats ------
        float mt = d0[0];
#pragma unroll
        for (int r = 0; r < 8; ++r) { mt = fmaxf(mt, d0[r]); mt = fmaxf(mt, d1[r]); }
        mt = fmaxf(mt, __shfl_xor(mt, 16, 32));
        const float mn = fmaxf(m, mt);
        float p0[8], p1[8], ls = 0.f;
#pragma unroll
        for (int r = 0; r < 8; ++r) {
            p0[r] = __expf(d0[r] - mn);
            p1[r] = __expf(d1[r] - mn);
            ls += p0[r] + p1[r];
        }
        ls += __shfl_xor(ls, 16, 32);
        const float alpha = __expf(m - mn);
        l = l * alpha + ls;
        m = mn;
#pragma unroll
        for (int ct = 0; ct < 4; ++ct) acc[ct] = acc[ct] * alpha;

        // ---- assemble P as B-operand (K-dim = s) via one xor-16 exchange ---
        unsigned pk0[4], pk1[4];
#pragma unroll
        for (int i = 0; i < 4; ++i) {
            pk0[i] = pack2(p0[2 * i], p0[2 * i + 1]);
            pk1[i] = pack2(p1[2 * i], p1[2 * i + 1]);
        }
        union BU { v16bf v; unsigned u[8]; } bfr;
#pragma unroll
        for (int i = 0; i < 4; ++i) {
            unsigned x = hi ? pk0[i] : pk1[i];
            unsigned y = (unsigned)__shfl_xor((int)x, 16, 32);
            bfr.u[i]     = hi ? y : pk0[i];
            bfr.u[4 + i] = hi ? pk1[i] : y;
        }

        // ---- O(c,t) += V(16c x 32s) x P(32s x 16t) -------------------------
#pragma unroll
        for (int ct = 0; ct < 4; ++ct) {
            U16 va;
            va.h[0] = *(const v8bf*)&vT[cur][16 * ct + ln][16 * hi];
            va.h[1] = *(const v8bf*)&vT[cur][16 * ct + ln][16 * hi + 8];
            acc[ct] = __builtin_amdgcn_wmma_f32_16x16x32_bf16(false, va.v, false, bfr.v, (short)0, acc[ct], false, false);
        }

        // Next tile's copy must be complete (and this tile's readers done)
        // before the buffers are reused.
        wait_async0();
        __syncthreads();
    }

    // ---- epilogue: out[(bh*64 + c)*T + t] = acc / l ------------------------
    const float inv = 1.0f / l;
    float* op = out + (size_t)bh * DH * SEQ_T + tq;
#pragma unroll
    for (int ct = 0; ct < 4; ++ct)
#pragma unroll
        for (int r = 0; r < 8; ++r)
            op[(size_t)(16 * ct + 8 * hi + r) * SEQ_T] = acc[ct][r] * inv;
}

extern "C" void kernel_launch(void* const* d_in, const int* in_sizes, int n_in,
                              void* d_out, int out_size, void* d_ws, size_t ws_size,
                              hipStream_t stream) {
    const float* qkv = (const float*)d_in[0];
    float* out = (float*)d_out;
    const size_t planeElems = (size_t)NBH * SEQ_T * DH; // 4,194,304
    __bf16* Qb = (__bf16*)d_ws;
    __bf16* Kb = Qb + planeElems;
    __bf16* Vb = Kb + planeElems;

    dim3 gT(NBH * (SEQ_T / 64), 2);
    qk_transpose<<<gT, 256, 0, stream>>>(qkv, Qb, Kb);
    v_convert<<<(int)(planeElems / 4 / 256), 256, 0, stream>>>(qkv, Vb);

    dim3 g(SEQ_T / 128, NBH);
    flash_attn<<<g, 256, 0, stream>>>(Qb, Kb, Vb, out);
}
